// GraphConvolution_4045859193006
// MI455X (gfx1250) — matrix operand order repaired and checked by reading, not compiled, not run on hardware
//
#include <hip/hip_runtime.h>
#include <math.h>
#include <stdint.h>

// ---------------------------------------------------------------------------
// GCN attention network, MI455X (gfx1250, wave32).
// Heavy math: tiled bf16 WMMA GEMM (v_wmma_f32_16x16x32_bf16), double-buffered
// LDS staging via async global->LDS copies (ASYNCcnt / s_wait_asynccnt).
// ---------------------------------------------------------------------------

typedef __bf16 v16bf __attribute__((ext_vector_type(16)));
typedef __bf16 v8bf  __attribute__((ext_vector_type(8)));
typedef float  v8f   __attribute__((ext_vector_type(8)));

#define NN   1024          // nodes
#define BB   16            // batch (graphs)
#define FF   256           // feature dim
#define M1   (NN*BB)       // 16384 rows of the [N,B,F] flattening

// ---- gfx1250 async global->LDS copy: one 16B transfer per lane ------------
static __device__ __forceinline__ void async_b128(uint32_t lds_byte_addr,
                                                  const void* gaddr) {
    asm volatile("global_load_async_to_lds_b128 %0, %1, off"
                 :: "v"(lds_byte_addr), "v"((uint64_t)(uintptr_t)gaddr)
                 : "memory");
}
static __device__ __forceinline__ void wait_async0() {
    asm volatile("s_wait_asynccnt 0x0" ::: "memory");
}

// ---------------- fp32 -> bf16 conversion (vectorized x4) ------------------
__global__ __launch_bounds__(256) void k_f32_to_bf16(const float* __restrict__ src,
                                                     __bf16* __restrict__ dst, int n4) {
    int i = blockIdx.x * blockDim.x + threadIdx.x;
    if (i >= n4) return;
    float4 v = reinterpret_cast<const float4*>(src)[i];
    __bf16 o0 = (__bf16)v.x, o1 = (__bf16)v.y, o2 = (__bf16)v.z, o3 = (__bf16)v.w;
    ushort4 pack;
    pack.x = __builtin_bit_cast(unsigned short, o0);
    pack.y = __builtin_bit_cast(unsigned short, o1);
    pack.z = __builtin_bit_cast(unsigned short, o2);
    pack.w = __builtin_bit_cast(unsigned short, o3);
    reinterpret_cast<ushort4*>(dst)[i] = pack;
}

// ---------------- tiled bf16 WMMA GEMM -------------------------------------
// C[M,N] = A[M,K] @ op(B)  (+bias[N]) ; op(B)=B (row-major KxN) or B^T (transB)
// Block tile 128x128, K-step 32 (= one v_wmma_f32_16x16x32_bf16 K).
// 256 threads = 8 waves in a 2x4 grid; wave tile 64x32 = 4x2 WMMA tiles.
// Double-buffered LDS; A (and B when transB) staged with async-to-LDS.
// Requires M%128==0, N%128==0, K%32==0 (true for every GEMM in this net).
struct GemmP {
    const __bf16* A; const __bf16* B; const float* bias;
    float* Cf; __bf16* Ch;
    int lda, ldb, ldc, M, N, K;
    int transB, relu;
};

#define BM 128
#define BN 128
#define BK 32
#define BKP 48   // padded LDS row stride (elements): 96B rows, 16B-aligned

__global__ __launch_bounds__(256) void k_gemm_wmma(GemmP p) {
    __shared__ __bf16 As[2][BM * BKP];
    __shared__ __bf16 Bs[2][BN * BKP];

    const int tid  = threadIdx.x;
    const int wid  = tid >> 5;
    const int lane = tid & 31;
    const int wr   = wid >> 2;   // 0..1  (wave row -> 64 rows of M)
    const int wc   = wid & 3;    // 0..3  (wave col -> 32 cols of N)
    const int half = lane >> 4;  // 0..1
    const int r16  = lane & 15;

    const int m0 = blockIdx.y * BM;
    const int n0 = blockIdx.x * BN;

    // stage A tile [128 rows][32 K] (K contiguous): 2 async 16B ops / thread
    auto stageA = [&](int k0, __bf16* dst) {
        const int row = tid >> 1, h = tid & 1;
        const __bf16* g = p.A + (size_t)(m0 + row) * p.lda + k0 + h * 16;
        const uint32_t l = (uint32_t)(uintptr_t)&dst[row * BKP + h * 16];
        async_b128(l, g);
        async_b128(l + 16, g + 8);
    };
    // stage B^T tile [128 N][32 K] (K contiguous): async, rows are K-contig
    auto stageBt = [&](int k0, __bf16* dst) {
        const int n = tid >> 1, h = tid & 1;
        const __bf16* g = p.B + (size_t)(n0 + n) * p.ldb + k0 + h * 16;
        const uint32_t l = (uint32_t)(uintptr_t)&dst[n * BKP + h * 16];
        async_b128(l, g);
        async_b128(l + 16, g + 8);
    };
    // stage row-major B with transpose into [128 N][32 K]: sync scatter path
    auto stageBs = [&](int k0, __bf16* dst) {
        const int kk = tid >> 3, g8 = tid & 7;       // kk 0..31, 16 n's each
        const __bf16* g = p.B + (size_t)(k0 + kk) * p.ldb + n0 + g8 * 16;
        union { float4 f[2]; __bf16 e[16]; } lb;
        const float4* g4 = reinterpret_cast<const float4*>(g);
        lb.f[0] = g4[0];
        lb.f[1] = g4[1];
        #pragma unroll
        for (int j = 0; j < 16; ++j)
            dst[(g8 * 16 + j) * BKP + kk] = lb.e[j];
        if (k0 + BK < p.K)
            __builtin_prefetch(g + BK * p.ldb, 0, 1);
    };
    auto stage = [&](int k0, int buf) {
        stageA(k0, As[buf]);
        if (p.transB) stageBt(k0, Bs[buf]);
        else          stageBs(k0, Bs[buf]);
    };

    v8f acc[4][2] = {};

    const int nk = p.K / BK;
    stage(0, 0);                                   // prologue: tile 0 -> buf 0

    for (int i = 0; i < nk; ++i) {
        wait_async0();                             // this wave's tile-i data in LDS
        __syncthreads();                           // everyone's data in LDS
        if (i + 1 < nk) stage((i + 1) * BK, (i + 1) & 1);  // overlap next tile

        const __bf16* Ab = As[i & 1];
        const __bf16* Bb = Bs[i & 1];

        // ---- fragments per documented wave32 layouts ----
        union FA { v16bf v; v8bf h[2]; };
        FA afr[4], bfr[2];
        #pragma unroll
        for (int mt = 0; mt < 4; ++mt) {
            const int base = (wr * 64 + mt * 16 + r16) * BKP;
            afr[mt].h[0] = *reinterpret_cast<const v8bf*>(&Ab[base + half * 8]);
            afr[mt].h[1] = *reinterpret_cast<const v8bf*>(&Ab[base + 16 + half * 8]);
        }
        #pragma unroll
        for (int nt = 0; nt < 2; ++nt) {
            const int base = (wc * 32 + nt * 16 + r16) * BKP + half * 8;
            bfr[nt].h[0] = *reinterpret_cast<const v8bf*>(&Bb[base]);
            bfr[nt].h[1] = *reinterpret_cast<const v8bf*>(&Bb[base + 16]);
        }

        // ---- 8 WMMAs per wave per K-step ----
        #pragma unroll
        for (int mt = 0; mt < 4; ++mt)
            #pragma unroll
            for (int nt = 0; nt < 2; ++nt)
                acc[mt][nt] = __builtin_amdgcn_wmma_f32_16x16x32_bf16(
                    false, afr[mt].v, false, bfr[nt].v,
                    (short)0, acc[mt][nt], false, false);
    }

    // ---- epilogue: C/D layout => row = half*8 + j, col = lane&15 ----
    #pragma unroll
    for (int mt = 0; mt < 4; ++mt) {
        #pragma unroll
        for (int nt = 0; nt < 2; ++nt) {
            const int gn = n0 + wc * 32 + nt * 16 + r16;
            #pragma unroll
            for (int j = 0; j < 8; ++j) {
                const int gm = m0 + wr * 64 + mt * 16 + half * 8 + j;
                float v = acc[mt][nt][j];
                if (p.bias) v += p.bias[gn];
                if (p.relu) v = fmaxf(v, 0.0f);
                const size_t o = (size_t)gm * p.ldc + gn;
                if (p.Cf) p.Cf[o] = v;
                if (p.Ch) p.Ch[o] = (__bf16)v;
            }
        }
    }
}

// ---------------- softmax over rows of length 1024, bf16 out ---------------
__global__ __launch_bounds__(256) void k_softmax1024(const float* __restrict__ in,
                                                     __bf16* __restrict__ out) {
    __shared__ float red[256];
    const int row = blockIdx.x, t = threadIdx.x;
    const float* r = in + (size_t)row * 1024;
    float v[4], m = -INFINITY;
    #pragma unroll
    for (int i = 0; i < 4; ++i) { v[i] = r[t + i * 256]; m = fmaxf(m, v[i]); }
    red[t] = m; __syncthreads();
    for (int s = 128; s > 0; s >>= 1) { if (t < s) red[t] = fmaxf(red[t], red[t + s]); __syncthreads(); }
    m = red[0]; __syncthreads();
    float sum = 0.f;
    #pragma unroll
    for (int i = 0; i < 4; ++i) { v[i] = __expf(v[i] - m); sum += v[i]; }
    red[t] = sum; __syncthreads();
    for (int s = 128; s > 0; s >>= 1) { if (t < s) red[t] += red[t + s]; __syncthreads(); }
    const float inv = 1.0f / red[0];
    #pragma unroll
    for (int i = 0; i < 4; ++i) out[(size_t)row * 1024 + t + i * 256] = (__bf16)(v[i] * inv);
}

// ---------------- fused relu + sigmoid highway -----------------------------
// out[n,b,f] = relu(q+v+s0+s1)[b,n,f]*t + x[n,b,f]*(1-t), t = sigmoid(hw[n,b,f])
__global__ __launch_bounds__(256) void k_combine(const float* __restrict__ qf,
                                                 const float* __restrict__ vf,
                                                 const float* __restrict__ s0f,
                                                 const float* __restrict__ s1f,
                                                 const float* __restrict__ hwf,
                                                 const float* __restrict__ xf,
                                                 float* __restrict__ yf,
                                                 __bf16* __restrict__ ybf) {
    const int idx = blockIdx.x * blockDim.x + threadIdx.x;
    if (idx >= M1 * FF) return;
    const int f = idx & (FF - 1);
    const int m = idx >> 8;          // m = n*BB + b
    const int b = m & (BB - 1);
    const int n = m >> 4;
    const size_t bnf = ((size_t)b * NN + n) * FF + f;
    float o = qf[bnf] + vf[bnf] + s0f[bnf] + s1f[bnf];
    o = fmaxf(o, 0.0f);
    const float t = 1.0f / (1.0f + __expf(-hwf[idx]));
    const float x = xf[idx];
    const float y = o * t + x * (1.0f - t);
    yf[idx] = y;
    if (ybf) ybf[idx] = (__bf16)y;
}

// ---------------------------------------------------------------------------
static inline void cvt(hipStream_t s, const float* src, __bf16* dst, size_t n) {
    int n4 = (int)(n / 4);
    k_f32_to_bf16<<<(n4 + 255) / 256, 256, 0, s>>>(src, dst, n4);
}
static inline void gemm(hipStream_t s, const __bf16* A, int lda,
                        const __bf16* B, int ldb, int transB,
                        const float* bias, float* Cf, __bf16* Ch, int ldc,
                        int M, int N, int K, int relu) {
    GemmP p{A, B, bias, Cf, Ch, lda, ldb, ldc, M, N, K, transB, relu};
    dim3 g(N / BN, M / BM);
    k_gemm_wmma<<<g, 256, 0, s>>>(p);
}

extern "C" void kernel_launch(void* const* d_in, const int* in_sizes, int n_in,
                              void* d_out, int out_size, void* d_ws, size_t ws_size,
                              hipStream_t stream) {
    const float* f_inputs = (const float*)d_in[0];   // [N,B,F]
    const float* f_adj    = (const float*)d_in[1];   // [3,B,N,N]

    // ---- bump allocator over workspace ----
    char* ws = (char*)d_ws; size_t off = 0;
    auto alloc = [&](size_t bytes) { void* p = ws + off; off += (bytes + 255) & ~(size_t)255; return p; };

    const size_t NF  = (size_t)NN * FF;
    __bf16* adj_bf  = (__bf16*)alloc((size_t)3 * BB * NN * NN * 2);
    __bf16* x_bf    = (__bf16*)alloc((size_t)M1 * FF * 2);
    __bf16* y_bf    = (__bf16*)alloc((size_t)M1 * FF * 2);
    __bf16* supp_bf = (__bf16*)alloc((size_t)M1 * FF * 2);      // reused per k
    __bf16* kqv_bf  = (__bf16*)alloc((size_t)3 * BB * NF * 2);  // [k][b][N][F]
    float*  qv_f    = (float*) alloc((size_t)2 * BB * NF * 4);  // q,v fp32
    float*  logits  = (float*) alloc((size_t)BB * NN * NN * 4); // reused a0/a1
    __bf16* a_bf    = (__bf16*)alloc((size_t)BB * NN * NN * 2); // reused a0/a1
    __bf16* spre_bf = (__bf16*)alloc((size_t)BB * NF * 2);      // reused s0/s1
    float*  s0_f    = (float*) alloc((size_t)BB * NF * 4);
    float*  s1_f    = (float*) alloc((size_t)BB * NF * 4);
    float*  hw_f    = (float*) alloc((size_t)M1 * FF * 4);
    float*  y1_f    = (float*) alloc((size_t)M1 * FF * 4);      // layer-1 output [N,B,F]
    __bf16* w_bf    = (__bf16*)alloc((size_t)6 * FF * FF * 2);  // weight[3], w0, w1, wh

    // ---- one-time conversions ----
    cvt(stream, f_adj, adj_bf, (size_t)3 * BB * NN * NN);
    cvt(stream, f_inputs, x_bf, (size_t)M1 * FF);

    for (int l = 0; l < 2; ++l) {
        const int base = 2 + l * 8;
        const float* p_weight = (const float*)d_in[base + 0]; // [3,F,F]
        const float* p_bias   = (const float*)d_in[base + 1]; // [3,1,F]
        const float* p_w0     = (const float*)d_in[base + 2];
        const float* p_b0     = (const float*)d_in[base + 3];
        const float* p_w1     = (const float*)d_in[base + 4];
        const float* p_b1     = (const float*)d_in[base + 5];
        const float* p_wh     = (const float*)d_in[base + 6];
        const float* p_bh     = (const float*)d_in[base + 7];

        const float*  xf  = (l == 0) ? f_inputs : y1_f;
        const __bf16* xbf = (l == 0) ? x_bf : y_bf;

        __bf16* wk_bf = w_bf;                     // 3*F*F
        __bf16* w0_bf = w_bf + (size_t)3 * FF * FF;
        __bf16* w1_bf = w_bf + (size_t)4 * FF * FF;
        __bf16* wh_bf = w_bf + (size_t)5 * FF * FF;
        cvt(stream, p_weight, wk_bf, (size_t)3 * FF * FF);
        cvt(stream, p_w0, w0_bf, (size_t)FF * FF);
        cvt(stream, p_w1, w1_bf, (size_t)FF * FF);
        cvt(stream, p_wh, wh_bf, (size_t)FF * FF);

        // support_k = X @ W_k + bias_k ; then agg per batch: adj_k[b] @ support
        for (int k = 0; k < 3; ++k) {
            gemm(stream, xbf, FF, wk_bf + (size_t)k * FF * FF, FF, 0,
                 p_bias + (size_t)k * FF, nullptr, supp_bf, FF, M1, FF, FF, 0);
            for (int b = 0; b < BB; ++b) {
                float* cf = (k == 1) ? qv_f + (size_t)b * NF
                          : (k == 2) ? qv_f + (size_t)(BB + b) * NF : nullptr;
                gemm(stream,
                     adj_bf + ((size_t)k * BB + b) * NN * NN, NN,  // A = adj[k][b]
                     supp_bf + (size_t)b * FF, BB * FF, 0,          // B = support view
                     nullptr, cf, kqv_bf + ((size_t)k * BB + b) * NF, FF,
                     NN, FF, NN, 0);
            }
        }

        const __bf16* kb = kqv_bf;                        // [b][N][F]
        const __bf16* qb = kqv_bf + (size_t)BB * NF;
        const __bf16* vb = kqv_bf + (size_t)2 * BB * NF;
        const float*  qf = qv_f;
        const float*  vf = qv_f + (size_t)BB * NF;

        // attention 0: softmax(q@k^T) @ k @ w0^T + b0
        for (int b = 0; b < BB; ++b)
            gemm(stream, qb + (size_t)b * NF, FF, kb + (size_t)b * NF, FF, 1,
                 nullptr, logits + (size_t)b * NN * NN, nullptr, NN, NN, NN, FF, 0);
        k_softmax1024<<<BB * NN, 256, 0, stream>>>(logits, a_bf);
        for (int b = 0; b < BB; ++b)
            gemm(stream, a_bf + (size_t)b * NN * NN, NN, kb + (size_t)b * NF, FF, 0,
                 nullptr, nullptr, spre_bf + (size_t)b * NF, FF, NN, FF, NN, 0);
        gemm(stream, spre_bf, FF, w0_bf, FF, 1, p_b0, s0_f, nullptr, FF,
             BB * NN, FF, FF, 0);

        // attention 1: softmax(v@k^T) @ k @ w1^T + b1
        for (int b = 0; b < BB; ++b)
            gemm(stream, vb + (size_t)b * NF, FF, kb + (size_t)b * NF, FF, 1,
                 nullptr, logits + (size_t)b * NN * NN, nullptr, NN, NN, NN, FF, 0);
        k_softmax1024<<<BB * NN, 256, 0, stream>>>(logits, a_bf);
        for (int b = 0; b < BB; ++b)
            gemm(stream, a_bf + (size_t)b * NN * NN, NN, kb + (size_t)b * NF, FF, 0,
                 nullptr, nullptr, spre_bf + (size_t)b * NF, FF, NN, FF, NN, 0);
        gemm(stream, spre_bf, FF, w1_bf, FF, 1, p_b1, s1_f, nullptr, FF,
             BB * NN, FF, FF, 0);

        // highway gate pre-activation: X @ wh + bh
        gemm(stream, xbf, FF, wh_bf, FF, 0, p_bh, hw_f, nullptr, FF, M1, FF, FF, 0);

        // combine: relu(q+v+s0+s1), sigmoid highway, permute to [N,B,F]
        float*  yf  = (l == 0) ? y1_f : (float*)d_out;
        __bf16* ybf = (l == 0) ? y_bf : nullptr;
        k_combine<<<(M1 * FF + 255) / 256, 256, 0, stream>>>(
            qf, vf, s0_f, s1_f, hw_f, xf, yf, ybf);
    }
    (void)in_sizes; (void)n_in; (void)out_size; (void)ws_size;
}
